// GAU_78254304133894
// MI455X (gfx1250) — matrix operand order repaired and checked by reading, not compiled, not run on hardware
//
#include <hip/hip_runtime.h>
#include <hip/hip_bf16.h>

#define H_   512
#define E_   1024
#define S_   128
#define NCTX 1024
#define B_   32
#define TOK  (B_ * NCTX)      // 32768
#define UV_  (2 * E_ + S_)    // 2176

typedef __attribute__((ext_vector_type(16))) __bf16 bf16x16;
typedef __attribute__((ext_vector_type(8)))  __bf16 bf16x8;
typedef __attribute__((ext_vector_type(8)))  float  f32x8;

union Frag { bf16x16 v; bf16x8 h[2]; };

__device__ __forceinline__ __bf16 f2bf(float f) {
    unsigned u = __builtin_bit_cast(unsigned, f);
    unsigned r = u + 0x7FFFu + ((u >> 16) & 1u);   // round-to-nearest-even
    unsigned short h = (unsigned short)(r >> 16);
    return __builtin_bit_cast(__bf16, h);
}
__device__ __forceinline__ float bf2f(__bf16 b) {
    unsigned short h = __builtin_bit_cast(unsigned short, b);
    unsigned u = ((unsigned)h) << 16;
    return __builtin_bit_cast(float, u);
}

// ---------------- per-token scale_norm -> bf16 ----------------
__global__ void k_norm(const float* __restrict__ x, const float* __restrict__ g,
                       __bf16* __restrict__ xn) {
    int t = blockIdx.x;
    int tid = threadIdx.x;
    const float* xr = x + (size_t)t * H_;
    float a0 = xr[tid], a1 = xr[tid + 256];
    float ss = a0 * a0 + a1 * a1;
    for (int off = 16; off; off >>= 1) ss += __shfl_xor(ss, off, 32);
    __shared__ float red[8];
    int wid = tid >> 5, lane = tid & 31;
    if (lane == 0) red[wid] = ss;
    __syncthreads();
    if (tid == 0) { float s = 0.f; for (int i = 0; i < 8; ++i) s += red[i]; red[0] = s; }
    __syncthreads();
    float norm = sqrtf(red[0] * (1.0f / H_));
    float scale = g[0] / fmaxf(norm, 1e-5f);
    __bf16* xo = xn + (size_t)t * H_;
    xo[tid]       = f2bf(a0 * scale);
    xo[tid + 256] = f2bf(a1 * scale);
}

// ---------------- weight transpose + cast ----------------
__global__ void k_wuvT(const float* __restrict__ W, __bf16* __restrict__ WT) {
    int idx = blockIdx.x * 256 + threadIdx.x;          // WT[UV_][H_]
    if (idx >= UV_ * H_) return;
    int n = idx / H_, k = idx % H_;
    WT[idx] = f2bf(W[(size_t)k * UV_ + n]);
}
__global__ void k_woT(const float* __restrict__ W, __bf16* __restrict__ WT) {
    int idx = blockIdx.x * 256 + threadIdx.x;          // WT[H_][E_]
    if (idx >= H_ * E_) return;
    int hcol = idx / E_, e = idx % E_;
    WT[idx] = f2bf(W[(size_t)e * H_ + hcol]);
}

// ---- GEMM1: uv = silu(xn @ Wuv + buv); u row-major, v TRANSPOSED per batch, s ----
__global__ __launch_bounds__(256) void k_gemm1(
    const __bf16* __restrict__ Xn, const __bf16* __restrict__ WT,
    const float* __restrict__ buv,
    __bf16* __restrict__ U,           // [TOK][E_]
    __bf16* __restrict__ VT,          // [B_][E_][NCTX]
    __bf16* __restrict__ Sb) {        // [TOK][S_]
    int bid = blockIdx.x;
    int nt = bid % 34, mt = bid / 34;
    int m0 = mt * 128, n0 = nt * 64;
    int tid = threadIdx.x, wid = tid >> 5, lane = tid & 31;
    int waveM = wid >> 1, waveN = wid & 1;
    int lLo = lane & 15, lHi = lane >> 4;

    f32x8 acc[2][2] = {};
    int rowA0 = m0 + waveM * 32 + lLo;
    int colB0 = n0 + waveN * 32 + lLo;

    for (int kk = 0; kk < 16; ++kk) {
        Frag a[2], b[2];
        int kbA = kk * 32 + lHi * 8;
        int kbB = kk * 32 + lHi * 8;
        for (int mi = 0; mi < 2; ++mi) {
            const __bf16* p = Xn + (size_t)(rowA0 + mi * 16) * H_ + kbA;
            a[mi].h[0] = *(const bf16x8*)(p);
            a[mi].h[1] = *(const bf16x8*)(p + 16);
        }
        for (int ni = 0; ni < 2; ++ni) {
            const __bf16* p = WT + (size_t)(colB0 + ni * 16) * H_ + kbB;
            b[ni].h[0] = *(const bf16x8*)(p);
            b[ni].h[1] = *(const bf16x8*)(p + 16);
        }
        for (int mi = 0; mi < 2; ++mi)
            for (int ni = 0; ni < 2; ++ni)
                acc[mi][ni] = __builtin_amdgcn_wmma_f32_16x16x32_bf16(
                    false, a[mi].v, false, b[ni].v, (short)0, acc[mi][ni], false, false);
    }
    for (int mi = 0; mi < 2; ++mi)
        for (int ni = 0; ni < 2; ++ni) {
            int col = colB0 + ni * 16;
            float bias = buv[col];
            int rowb = m0 + waveM * 32 + mi * 16 + lHi * 8;   // 8 consecutive tokens
            float sv[8];
            for (int r = 0; r < 8; ++r) {
                float vv = acc[mi][ni][r] + bias;
                sv[r] = vv / (1.0f + __expf(-vv));             // silu
            }
            if (col < E_) {
                for (int r = 0; r < 8; ++r)
                    U[(size_t)(rowb + r) * E_ + col] = f2bf(sv[r]);
            } else if (col < 2 * E_) {
                int e = col - E_;
                int bb = rowb >> 10, nn = rowb & (NCTX - 1);
                bf16x8 pack;
                for (int r = 0; r < 8; ++r) pack[r] = f2bf(sv[r]);
                *(bf16x8*)(VT + ((size_t)bb * E_ + e) * NCTX + nn) = pack;
            } else {
                for (int r = 0; r < 8; ++r)
                    Sb[(size_t)(rowb + r) * S_ + (col - 2 * E_)] = f2bf(sv[r]);
            }
        }
}

// ---------------- gamma/beta + RoPE -> q,k ----------------
__global__ void k_rope(const __bf16* __restrict__ Sb, const float* __restrict__ gamma,
                       const float* __restrict__ beta,
                       __bf16* __restrict__ Q, __bf16* __restrict__ K) {
    int t = blockIdx.x;
    int i = threadIdx.x;                 // 0..63
    int n = t & (NCTX - 1);
    float s1 = bf2f(Sb[(size_t)t * S_ + i]);
    float s2 = bf2f(Sb[(size_t)t * S_ + i + 64]);
    float invf = __powf(10000.0f, (float)i * (1.0f / 64.0f));
    float ang = (float)n * invf;
    float sn = __sinf(ang), cs = __cosf(ang);
    for (int j = 0; j < 2; ++j) {
        float x1 = s1 * gamma[j * S_ + i]      + beta[j * S_ + i];
        float x2 = s2 * gamma[j * S_ + i + 64] + beta[j * S_ + i + 64];
        __bf16* dst = (j == 0) ? Q : K;
        dst[(size_t)t * S_ + i]      = f2bf(x1 * cs - x2 * sn);
        dst[(size_t)t * S_ + i + 64] = f2bf(x2 * cs + x1 * sn);
    }
}

// ---- QK: Sqk[b][n][m] = relu(q.k / sqrtS)^2 in bf16 (computed exactly once) ----
__global__ __launch_bounds__(256) void k_qk(
    const __bf16* __restrict__ Q, const __bf16* __restrict__ Kq,
    __bf16* __restrict__ Sqk) {        // [B_][NCTX][NCTX]
    int bid = blockIdx.x;
    int mt2 = bid & 15, nt = (bid >> 4) & 15, b = bid >> 8;
    int n0 = nt * 64, m0 = mt2 * 64;
    int tid = threadIdx.x, wid = tid >> 5, lane = tid & 31;
    int waveM = wid >> 1, waveN = wid & 1;
    int lLo = lane & 15, lHi = lane >> 4;

    f32x8 acc[2][2] = {};
    int rowA0 = b * NCTX + n0 + waveM * 32 + lLo;   // q token rows
    int rowB0 = b * NCTX + m0 + waveN * 32 + lLo;   // k token rows (output cols)

    for (int kk = 0; kk < 4; ++kk) {                // K = 128
        Frag a[2], bfr[2];
        for (int mi = 0; mi < 2; ++mi) {
            const __bf16* p = Q + (size_t)(rowA0 + mi * 16) * S_ + kk * 32 + lHi * 8;
            a[mi].h[0] = *(const bf16x8*)(p);
            a[mi].h[1] = *(const bf16x8*)(p + 16);
        }
        for (int ni = 0; ni < 2; ++ni) {
            const __bf16* p = Kq + (size_t)(rowB0 + ni * 16) * S_ + kk * 32 + lHi * 8;
            bfr[ni].h[0] = *(const bf16x8*)(p);
            bfr[ni].h[1] = *(const bf16x8*)(p + 16);
        }
        for (int mi = 0; mi < 2; ++mi)
            for (int ni = 0; ni < 2; ++ni)
                acc[mi][ni] = __builtin_amdgcn_wmma_f32_16x16x32_bf16(
                    false, a[mi].v, false, bfr[ni].v, (short)0, acc[mi][ni], false, false);
    }
    const float qscale = 1.0f / (float)S_;          // (1/sqrt(S))^2
    __bf16* srow = Sqk + ((size_t)b << 20);
    for (int mi = 0; mi < 2; ++mi)
        for (int ni = 0; ni < 2; ++ni) {
            int mcol = m0 + waveN * 32 + ni * 16 + lLo;
            for (int r = 0; r < 8; ++r) {
                int nrow = n0 + waveM * 32 + mi * 16 + lHi * 8 + r;
                float d = fmaxf(acc[mi][ni][r], 0.0f);
                srow[(size_t)nrow * NCTX + mcol] = f2bf(d * d * qscale);
            }
        }
}

// ---- PV: y = u * (Sqk @ v)  (in place over U), per-batch B from VT ----
__global__ __launch_bounds__(256) void k_pv(
    const __bf16* __restrict__ Sqk, const __bf16* __restrict__ VT,
    __bf16* __restrict__ U) {
    int bid = blockIdx.x;
    int et = bid & 15, mt = bid >> 4;
    int m0 = mt * 128, e0 = et * 64;     // m0: token rows (128 within one batch)
    int b = m0 >> 10;
    int tid = threadIdx.x, wid = tid >> 5, lane = tid & 31;
    int waveM = wid >> 1, waveN = wid & 1;
    int lLo = lane & 15, lHi = lane >> 4;

    f32x8 acc[2][2] = {};
    int rowA0 = m0 + waveM * 32 + lLo;
    int colB0 = e0 + waveN * 32 + lLo;
    const __bf16* vtb = VT + ((size_t)b * E_) * NCTX;

    for (int kk = 0; kk < 32; ++kk) {    // K = NCTX = 1024
        Frag a[2], bfr[2];
        for (int mi = 0; mi < 2; ++mi) {
            const __bf16* p = Sqk + (size_t)(rowA0 + mi * 16) * NCTX + kk * 32 + lHi * 8;
            a[mi].h[0] = *(const bf16x8*)(p);
            a[mi].h[1] = *(const bf16x8*)(p + 16);
        }
        for (int ni = 0; ni < 2; ++ni) {
            const __bf16* p = vtb + (size_t)(colB0 + ni * 16) * NCTX + kk * 32 + lHi * 8;
            bfr[ni].h[0] = *(const bf16x8*)(p);
            bfr[ni].h[1] = *(const bf16x8*)(p + 16);
        }
        for (int mi = 0; mi < 2; ++mi)
            for (int ni = 0; ni < 2; ++ni)
                acc[mi][ni] = __builtin_amdgcn_wmma_f32_16x16x32_bf16(
                    false, a[mi].v, false, bfr[ni].v, (short)0, acc[mi][ni], false, false);
    }
    for (int mi = 0; mi < 2; ++mi)
        for (int ni = 0; ni < 2; ++ni) {
            int col = colB0 + ni * 16;
            for (int r = 0; r < 8; ++r) {
                int row = m0 + waveM * 32 + mi * 16 + lHi * 8 + r;
                size_t idx = (size_t)row * E_ + col;
                float uval = bf2f(U[idx]);
                U[idx] = f2bf(uval * acc[mi][ni][r]);   // y overwrites u
            }
        }
}

// ---------------- GEMM3: out = y @ Wo + bo + x ----------------
__global__ __launch_bounds__(256) void k_gemm3(
    const __bf16* __restrict__ Y, const __bf16* __restrict__ WoT,
    const float* __restrict__ bo, const float* __restrict__ x,
    float* __restrict__ out) {
    int bid = blockIdx.x;
    int nt = bid & 7, mt = bid >> 3;
    int m0 = mt * 128, n0 = nt * 64;
    int tid = threadIdx.x, wid = tid >> 5, lane = tid & 31;
    int waveM = wid >> 1, waveN = wid & 1;
    int lLo = lane & 15, lHi = lane >> 4;
    f32x8 acc[2][2] = {};
    int rowA0 = m0 + waveM * 32 + lLo;
    int colB0 = n0 + waveN * 32 + lLo;
    for (int kk = 0; kk < 32; ++kk) {
        Frag a[2], b[2];
        for (int mi = 0; mi < 2; ++mi) {
            const __bf16* p = Y + (size_t)(rowA0 + mi * 16) * E_ + kk * 32 + lHi * 8;
            a[mi].h[0] = *(const bf16x8*)(p);
            a[mi].h[1] = *(const bf16x8*)(p + 16);
        }
        for (int ni = 0; ni < 2; ++ni) {
            const __bf16* p = WoT + (size_t)(colB0 + ni * 16) * E_ + kk * 32 + lHi * 8;
            b[ni].h[0] = *(const bf16x8*)(p);
            b[ni].h[1] = *(const bf16x8*)(p + 16);
        }
        for (int mi = 0; mi < 2; ++mi)
            for (int ni = 0; ni < 2; ++ni)
                acc[mi][ni] = __builtin_amdgcn_wmma_f32_16x16x32_bf16(
                    false, a[mi].v, false, b[ni].v, (short)0, acc[mi][ni], false, false);
    }
    for (int mi = 0; mi < 2; ++mi)
        for (int ni = 0; ni < 2; ++ni) {
            int col = colB0 + ni * 16;
            float bias = bo[col];
            for (int r = 0; r < 8; ++r) {
                int row = m0 + waveM * 32 + mi * 16 + lHi * 8 + r;
                out[(size_t)row * H_ + col] = acc[mi][ni][r] + bias + x[(size_t)row * H_ + col];
            }
        }
}

extern "C" void kernel_launch(void* const* d_in, const int* in_sizes, int n_in,
                              void* d_out, int out_size, void* d_ws, size_t ws_size,
                              hipStream_t stream) {
    const float* x     = (const float*)d_in[0];
    const float* Wuv   = (const float*)d_in[1];
    const float* buv   = (const float*)d_in[2];
    const float* gamma = (const float*)d_in[3];
    const float* beta  = (const float*)d_in[4];
    const float* Wo    = (const float*)d_in[5];
    const float* bo    = (const float*)d_in[6];
    const float* g     = (const float*)d_in[7];
    float* out = (float*)d_out;

    char* ws = (char*)d_ws;
    auto alloc = [&](size_t bytes) -> char* {
        char* p = ws; ws += (bytes + 255) & ~(size_t)255; return p;
    };
    __bf16* xn   = (__bf16*)alloc((size_t)TOK * H_ * 2);
    __bf16* WuvT = (__bf16*)alloc((size_t)UV_ * H_ * 2);
    __bf16* WoT  = (__bf16*)alloc((size_t)H_ * E_ * 2);
    __bf16* U    = (__bf16*)alloc((size_t)TOK * E_ * 2);
    __bf16* VT   = (__bf16*)alloc((size_t)TOK * E_ * 2);
    __bf16* Sb   = (__bf16*)alloc((size_t)TOK * S_ * 2);
    __bf16* Qb   = (__bf16*)alloc((size_t)TOK * S_ * 2);
    __bf16* Kb   = (__bf16*)alloc((size_t)TOK * S_ * 2);
    __bf16* Sqk  = (__bf16*)alloc((size_t)B_ * NCTX * NCTX * 2);
    (void)ws_size; (void)in_sizes; (void)n_in; (void)out_size;

    k_norm<<<TOK, 256, 0, stream>>>(x, g, xn);
    k_wuvT<<<(UV_ * H_ + 255) / 256, 256, 0, stream>>>(Wuv, WuvT);
    k_woT<<<(H_ * E_ + 255) / 256, 256, 0, stream>>>(Wo, WoT);
    k_gemm1<<<256 * 34, 256, 0, stream>>>(xn, WuvT, buv, U, VT, Sb);
    k_rope<<<TOK, 64, 0, stream>>>(Sb, gamma, beta, Qb, Kb);
    k_qk<<<B_ * 16 * 16, 256, 0, stream>>>(Qb, Kb, Sqk);
    k_pv<<<256 * 16, 256, 0, stream>>>(Sqk, VT, U);
    k_gemm3<<<256 * 8, 256, 0, stream>>>(U, WoT, bo, x, out);
}
